// Model_19688130085008
// MI455X (gfx1250) — matrix operand order repaired and checked by reading, not compiled, not run on hardware
//
#include <hip/hip_runtime.h>
#include <hip/hip_bf16.h>
#include <math.h>

// ---------------- problem constants (from reference) ----------------
#define NN      40000
#define EE      200000
#define IN_DIM  32
#define HID     64
#define HEADS   4
#define OUT_DIM 32
#define MAX_IT  8
#define KDIM    128      // 2*HID
#define ODIM    512      // fused xl|xr output columns (2 * HEADS*HID)
#define NSLOPE  0.2f

typedef __attribute__((ext_vector_type(16))) _Float16 v16h;
typedef __attribute__((ext_vector_type(8)))  float    v8f;

union FragU { v16h v; uint4 q[2]; };

// ---------------------------------------------------------------
// Pack W = [W_l ; W_r] (512 x 128, row-major f32) into WMMA B-fragment
// layout, f16.  B (32x16 per k-block) per-lane layout: column N = lane&15,
// K = kb*32 + (j>>3)*16 + (lane>>4)*8 + (j&7)  for j=0..15.
// wpack[((t*4+kb)*32 + lane)*16 + j]
// ---------------------------------------------------------------
__global__ void k_pack_w(const float* __restrict__ Wl,
                         const float* __restrict__ Wr,
                         _Float16* __restrict__ wpack) {
    int tid  = blockIdx.x * blockDim.x + threadIdx.x;   // 65536 threads
    int j    = tid & 15;
    int lane = (tid >> 4) & 31;
    int kb   = (tid >> 9) & 3;
    int t    = tid >> 11;                                // o-tile 0..31
    int o = t * 16 + (lane & 15);
    int k = kb * 32 + (j >> 3) * 16 + (lane >> 4) * 8 + (j & 7);
    float w = (o < 256) ? Wl[o * KDIM + k] : Wr[(o - 256) * KDIM + k];
    wpack[tid] = (_Float16)w;
}

// ---------------------------------------------------------------
// h0 = feat @ W_in^T + b_in ; initialize x = [h0 | h0] in f16
// ---------------------------------------------------------------
__global__ void k_init_h0(const float* __restrict__ feat,
                          const float* __restrict__ Win,
                          const float* __restrict__ bin,
                          _Float16* __restrict__ x) {
    int tid = blockIdx.x * blockDim.x + threadIdx.x;    // N*64
    int n = tid >> 6, o = tid & 63;
    float acc = bin[o];
    const float* f = feat + (size_t)n * IN_DIM;
    const float* w = Win + (size_t)o * IN_DIM;
#pragma unroll
    for (int k = 0; k < IN_DIM; ++k) acc = fmaf(f[k], w[k], acc);
    _Float16 hv = (_Float16)acc;
    x[(size_t)n * KDIM + o]      = hv;   // h  (iter 0: h == h0)
    x[(size_t)n * KDIM + 64 + o] = hv;   // h0 (fixed for all iterations)
}

// ---------------------------------------------------------------
// Fused GEMM: xlxr(N x 512) = x(N x 128, f16) @ [W_l;W_r]^T + [b_l;b_r]
// One wave computes a 16(M) x 16(O) tile, K=128 via 4x v_wmma_f32_16x16x32_f16.
// A-fragment (16-bit A 16x32): lane r=lane&15 -> row M=r;
//   oct=lane>>4: halves 0..7 = K kb*32+oct*8.. , halves 8..15 = +16.
// ---------------------------------------------------------------
__global__ void k_gemm(const _Float16* __restrict__ x,
                       const _Float16* __restrict__ wpack,
                       const float* __restrict__ bl,
                       const float* __restrict__ br,
                       float* __restrict__ xlxr) {
    const int wave = threadIdx.x >> 5;
    const int lane = threadIdx.x & 31;
    const int m0   = blockIdx.x * 16;            // node tile
    const int t    = blockIdx.y * 8 + wave;      // output col tile 0..31
    const int r    = lane & 15;
    const int oct  = lane >> 4;

    const _Float16* arow = x + (size_t)(m0 + r) * KDIM + oct * 8;
    v8f c = {0.f, 0.f, 0.f, 0.f, 0.f, 0.f, 0.f, 0.f};
#pragma unroll
    for (int kb = 0; kb < 4; ++kb) {
        FragU ua, ub;
        ua.q[0] = *(const uint4*)(arow + kb * 32);
        ua.q[1] = *(const uint4*)(arow + kb * 32 + 16);
        const _Float16* bptr = wpack + ((size_t)(t * 4 + kb) * 32 + lane) * 16;
        ub.q[0] = *(const uint4*)(bptr);
        ub.q[1] = *(const uint4*)(bptr + 8);
        c = __builtin_amdgcn_wmma_f32_16x16x32_f16(
                /*neg_a=*/false, ua.v, /*neg_b=*/false, ub.v,
                /*c_mod=*/(short)0, c, /*reuse_a=*/false, /*reuse_b=*/false);
    }
    // C/D layout: lanes 0-15 -> N=lane, M rows 0..7; lanes 16-31 -> M rows 8..15
    const int col  = t * 16 + r;
    const float bias = (col < 256) ? bl[col] : br[col - 256];
    const int row0 = m0 + oct * 8;
#pragma unroll
    for (int rr = 0; rr < 8; ++rr)
        xlxr[(size_t)(row0 + rr) * ODIM + col] = c[rr] + bias;
}

// ---------------------------------------------------------------
// Edge pass 1: logits -> w = active ? exp(logit) : 0 ; denom[dst] += w
// One wave per edge; lane d covers hid via d and d+32.
// ---------------------------------------------------------------
__global__ void k_edge_logits(const int* __restrict__ ei,
                              const int* __restrict__ emask,
                              const float* __restrict__ xlxr,
                              const float* __restrict__ att,
                              float* __restrict__ wedge,
                              float* __restrict__ denom,
                              int iter) {
    const int e    = blockIdx.x * 8 + (threadIdx.x >> 5);
    const int lane = threadIdx.x & 31;
    const int src = ei[e];
    const int dst = ei[EE + e];
    const bool active = iter < emask[e];
    const float* xls = xlxr + (size_t)src * ODIM;         // xl[src]
    const float* xrd = xlxr + (size_t)dst * ODIM + 256;   // xr[dst]
#pragma unroll
    for (int h = 0; h < HEADS; ++h) {
        const int d0 = h * HID + lane;
        float e0 = xls[d0]      + xrd[d0];
        float e1 = xls[d0 + 32] + xrd[d0 + 32];
        e0 = e0 > 0.f ? e0 : NSLOPE * e0;
        e1 = e1 > 0.f ? e1 : NSLOPE * e1;
        float p = e0 * att[d0] + e1 * att[d0 + 32];
#pragma unroll
        for (int off = 16; off > 0; off >>= 1) p += __shfl_xor(p, off, 32);
        if (lane == 0) {
            float w = active ? expf(p) : 0.f;
            wedge[(size_t)e * HEADS + h] = w;
            if (w != 0.f) atomicAdd(&denom[(size_t)dst * HEADS + h], w);
        }
    }
}

// ---------------------------------------------------------------
// Edge pass 2: out[dst] += xl[src] * (w / max(denom[dst],1e-16))
// ---------------------------------------------------------------
__global__ void k_edge_scatter(const int* __restrict__ ei,
                               const int* __restrict__ emask,
                               const float* __restrict__ xlxr,
                               const float* __restrict__ wedge,
                               const float* __restrict__ denom,
                               float* __restrict__ outacc,
                               int iter) {
    const int e    = blockIdx.x * 8 + (threadIdx.x >> 5);
    const int lane = threadIdx.x & 31;
    if (iter >= emask[e]) return;                 // wave-uniform branch
    const int src = ei[e];
    const int dst = ei[EE + e];
    const float* xls = xlxr + (size_t)src * ODIM;
    float* outd = outacc + (size_t)dst * (HEADS * HID);
#pragma unroll
    for (int h = 0; h < HEADS; ++h) {
        float w = wedge[(size_t)e * HEADS + h];
        float alpha = w / fmaxf(denom[(size_t)dst * HEADS + h], 1e-16f);
        const int d0 = h * HID + lane;
        atomicAdd(&outd[d0],      xls[d0]      * alpha);
        atomicAdd(&outd[d0 + 32], xls[d0 + 32] * alpha);
    }
}

// ---------------------------------------------------------------
// Node update: h = tanh(sum_heads(out + b_conv)); temp = h!=0 ? h : temp;
// refresh f16 x[:, 0:64]; re-zero out/denom in place for next iteration.
// ---------------------------------------------------------------
__global__ void k_node(const float* __restrict__ bconv,
                       float* __restrict__ outacc,
                       float* __restrict__ denom,
                       float* __restrict__ temp,
                       _Float16* __restrict__ x) {
    int tid = blockIdx.x * blockDim.x + threadIdx.x;     // N*64
    int n = tid >> 6, d = tid & 63;
    float* o = outacc + (size_t)n * (HEADS * HID);
    float s = bconv[d] + bconv[64 + d] + bconv[128 + d] + bconv[192 + d];
    s += o[d] + o[64 + d] + o[128 + d] + o[192 + d];
    o[d] = 0.f; o[64 + d] = 0.f; o[128 + d] = 0.f; o[192 + d] = 0.f;
    if (d < HEADS) denom[(size_t)n * HEADS + d] = 0.f;
    float h = tanhf(s);
    if (h != 0.f) temp[tid] = h;
    x[(size_t)n * KDIM + d] = (_Float16)h;
}

// ---------------------------------------------------------------
// Final projection: out = temp @ W_g^T + b_g   (N x 32)
// ---------------------------------------------------------------
__global__ void k_final(const float* __restrict__ temp,
                        const float* __restrict__ Wg,
                        const float* __restrict__ bg,
                        float* __restrict__ out) {
    int tid = blockIdx.x * blockDim.x + threadIdx.x;     // N*32
    int n = tid >> 5, o = tid & 31;
    float acc = bg[o];
    const float* t = temp + (size_t)n * HID;
    const float* w = Wg + (size_t)o * HID;
#pragma unroll
    for (int k = 0; k < HID; ++k) acc = fmaf(t[k], w[k], acc);
    out[tid] = acc;
}

// ---------------------------------------------------------------
extern "C" void kernel_launch(void* const* d_in, const int* in_sizes, int n_in,
                              void* d_out, int out_size, void* d_ws, size_t ws_size,
                              hipStream_t stream) {
    const float* feat  = (const float*)d_in[0];
    const int*   ei    = (const int*)  d_in[1];
    const int*   emask = (const int*)  d_in[2];
    // d_in[3] = current_state (all ones -> identity gather; unused)
    const float* Win   = (const float*)d_in[4];
    const float* bin   = (const float*)d_in[5];
    const float* Wl    = (const float*)d_in[6];
    const float* bl    = (const float*)d_in[7];
    const float* Wr    = (const float*)d_in[8];
    const float* br    = (const float*)d_in[9];
    const float* att   = (const float*)d_in[10];
    const float* bconv = (const float*)d_in[11];
    const float* Wg    = (const float*)d_in[12];
    const float* bg    = (const float*)d_in[13];

    // ---- workspace layout (bytes, 256-aligned) ----
    char* ws = (char*)d_ws;
    size_t off = 0;
    auto alloc = [&](size_t bytes) {
        size_t p = off; off = (off + bytes + 255) & ~(size_t)255; return p;
    };
    size_t off_x     = alloc((size_t)NN * KDIM * 2);          // f16 [h|h0]
    size_t off_wp    = alloc((size_t)ODIM * KDIM * 2);        // f16 packed W
    size_t off_xlxr  = alloc((size_t)NN * ODIM * 4);          // f32 xl|xr
    size_t off_wedge = alloc((size_t)EE * HEADS * 4);         // f32 per-edge w
    size_t off_denom = alloc((size_t)NN * HEADS * 4);         // f32 denom
    size_t off_out   = alloc((size_t)NN * HEADS * HID * 4);   // f32 out accum
    size_t off_temp  = alloc((size_t)NN * HID * 4);           // f32 temp
    if (off > ws_size) return;  // workspace too small: bail deterministically

    _Float16* x     = (_Float16*)(ws + off_x);
    _Float16* wpack = (_Float16*)(ws + off_wp);
    float* xlxr  = (float*)(ws + off_xlxr);
    float* wedge = (float*)(ws + off_wedge);
    float* denom = (float*)(ws + off_denom);
    float* outac = (float*)(ws + off_out);
    float* temp  = (float*)(ws + off_temp);

    // zero accumulators once (node kernel re-zeros out/denom each iteration)
    hipMemsetAsync(temp,  0, (size_t)NN * HID * 4, stream);
    hipMemsetAsync(denom, 0, (size_t)NN * HEADS * 4, stream);
    hipMemsetAsync(outac, 0, (size_t)NN * HEADS * HID * 4, stream);

    k_pack_w<<<(ODIM * KDIM) / 256, 256, 0, stream>>>(Wl, Wr, wpack);
    k_init_h0<<<(NN * HID) / 256, 256, 0, stream>>>(feat, Win, bin, x);

    for (int it = 0; it < MAX_IT; ++it) {
        dim3 g(NN / 16, 4);                       // 2500 x 4, 8 waves/block
        k_gemm<<<g, 256, 0, stream>>>(x, wpack, bl, br, xlxr);
        k_edge_logits<<<EE / 8, 256, 0, stream>>>(ei, emask, xlxr, att,
                                                  wedge, denom, it);
        k_edge_scatter<<<EE / 8, 256, 0, stream>>>(ei, emask, xlxr, wedge,
                                                   denom, outac, it);
        k_node<<<(NN * HID) / 256, 256, 0, stream>>>(bconv, outac, denom,
                                                     temp, x);
    }
    k_final<<<(NN * OUT_DIM) / 256, 256, 0, stream>>>(temp, Wg, bg,
                                                      (float*)d_out);
}
